// DownsampleExtractor_69518340653554
// MI455X (gfx1250) — matrix operand order repaired and checked by reading, not compiled, not run on hardware
//
#include <hip/hip_runtime.h>
#include <hip/hip_bf16.h>

// ---------------------------------------------------------------------------
// DownsampleExtractor for MI455X (gfx1250, wave32, WMMA).
//
// Roofline: K+V streams = 510 MB, weights ~101 MB, outputs ~2 MB -> ~613 MB
// @ 23.3 TB/s ~ 25us.  After folding Wk into the query (scores) and Wv past
// the attention (sum_t attn == 1), total math is ~3 GFLOP -> purely HBM bound.
// The two streaming kernels use v_wmma_f32_16x16x32_bf16; f32->bf16 uses the
// native converts via (__bf16) casts (v_cvt_pk_bf16_f32).
// ---------------------------------------------------------------------------

#define B_  8
#define L_  27
#define T_  256
#define D_  1152
#define G_  9
#define GS_ 3
#define DD_ 512
#define H_  8
#define HD_ 64
#define OD_ 2048
#define KCH 36            // D_/32 k-chunks for the scores GEMM
#define SCALE_ 0.125f     // 1/sqrt(64)

typedef __bf16  bf16_t;
typedef bf16_t  v16bf __attribute__((ext_vector_type(16)));
typedef float   v8f   __attribute__((ext_vector_type(8)));

union BF16x16 { v16bf v; bf16_t b[16]; uint4 q[2]; };

__device__ __forceinline__ bf16_t tobf(float x) { return (bf16_t)x; }

// ---------------------------------------------------------------------------
// Kernel 0: Qp[l, ff] = query[l,:] @ Wq[g] + bq[g]          (27 x 512, tiny)
// ---------------------------------------------------------------------------
__global__ __launch_bounds__(256) void k_qproj(const float* __restrict__ query,
                                               const float* __restrict__ Wq,
                                               const float* __restrict__ bq,
                                               float* __restrict__ Qp) {
    int idx = blockIdx.x * 256 + threadIdx.x;      // 27*512 = 13824
    if (idx >= L_ * DD_) return;
    int l = idx >> 9, ff = idx & 511, g = l / GS_;
    float acc = bq[g * DD_ + ff];
    const float* q = query + l * D_;
    const float* w = Wq + (long)g * D_ * DD_ + ff;
    for (int d = 0; d < D_; ++d) acc += q[d] * w[(long)d * DD_];
    Qp[idx] = acc;
}

// ---------------------------------------------------------------------------
// Kernel 1: q~ = SCALE * Wk[g] @ Qp (per head), emitted directly in the WMMA
// B-operand layout: qtB[(l*36 + c)*32 + lane][16 elems], bf16.
//   lane col n = lane&15 (head, >=8 zero-pad); elem j -> d = c*32+(lane>>4)*8+(j&7)+(j>>3)*16
// (bk is dropped: constant over t -> softmax-invariant.)
// ---------------------------------------------------------------------------
__global__ __launch_bounds__(256) void k_qtilde(const float* __restrict__ Qp,
                                                const float* __restrict__ Wk,
                                                bf16_t* __restrict__ qtB) {
    int idx = blockIdx.x * 256 + threadIdx.x;      // 27*36*32*16 = 497664
    if (idx >= L_ * KCH * 32 * 16) return;
    int j    = idx & 15;
    int lane = (idx >> 4) & 31;
    int rem  = idx % (KCH * 512);
    int c    = rem >> 9;
    int l    = idx / (KCH * 512);
    int n    = lane & 15;
    int d    = c * 32 + (lane >> 4) * 8 + (j & 7) + (j >> 3) * 16;
    int g    = l / GS_;
    float acc = 0.f;
    if (n < H_) {
        const float* qp = Qp + l * DD_ + n * HD_;
        const float* wk = Wk + ((long)g * D_ + d) * DD_ + n * HD_;
        for (int f = 0; f < HD_; ++f) acc += qp[f] * wk[f];
        acc *= SCALE_;
    }
    qtB[idx] = tobf(acc);
}

// ---------------------------------------------------------------------------
// Kernel 2: scores[b,l,h,t] = K[b,l,t,:] . q~[l,h,:]   (streams K: 255 MB)
// 1 wave per 16-row t-tile; 36 x v_wmma_f32_16x16x32_bf16 per tile.
// ---------------------------------------------------------------------------
__global__ __launch_bounds__(32) void k_scores(const float* __restrict__ K,
                                               const bf16_t* __restrict__ qtB,
                                               float* __restrict__ scores) {
    int blk = blockIdx.x;                 // (b*27+l)*16 + t-tile
    int bl  = blk >> 4;
    int tt  = blk & 15;
    int l   = bl % L_;
    int lane = threadIdx.x;
    int row  = lane & 15;                 // A-matrix row (t within tile)
    int hi   = lane >> 4;                 // K-offset selector
    const float* Kb = K + ((long)bl * T_ + tt * 16 + row) * D_;
    v8f acc = {};
    for (int c = 0; c < KCH; ++c) {
        const float* ap = Kb + c * 32 + hi * 8;
        float av[16];
        *(float4*)&av[0]  = *(const float4*)(ap);
        *(float4*)&av[4]  = *(const float4*)(ap + 4);
        *(float4*)&av[8]  = *(const float4*)(ap + 16);
        *(float4*)&av[12] = *(const float4*)(ap + 20);
        BF16x16 A;
        #pragma unroll
        for (int j = 0; j < 16; ++j) A.b[j] = tobf(av[j]);
        BF16x16 Bv;
        const uint4* bp = (const uint4*)(qtB + (((long)l * KCH + c) * 32 + lane) * 16);
        Bv.q[0] = bp[0]; Bv.q[1] = bp[1];
        acc = __builtin_amdgcn_wmma_f32_16x16x32_bf16(false, A.v, false, Bv.v,
                                                      (short)0, acc, false, false);
    }
    int h = lane & 15;
    #pragma unroll
    for (int r = 0; r < 8; ++r) {
        int m = r + hi * 8;               // t within tile
        if (h < H_)
            scores[((long)bl * H_ + h) * T_ + tt * 16 + m] = acc[r];
    }
}

// ---------------------------------------------------------------------------
// Kernel 3: softmax over t; writes attn directly in the WMMA A-operand layout
// attnA[(bl*8 + c)*32 + la][16] bf16, pad rows (m>=8) zeroed.
// ---------------------------------------------------------------------------
__global__ __launch_bounds__(256) void k_softmax(const float* __restrict__ scores,
                                                 bf16_t* __restrict__ attnA) {
    __shared__ float red[256];
    int blh = blockIdx.x;                 // (b*27+l)*8 + h
    long bl = blh / H_;
    int  h  = blh % H_;
    int  t  = threadIdx.x;
    float v = scores[(long)blh * T_ + t];
    red[t] = v; __syncthreads();
    for (int s = 128; s > 0; s >>= 1) { if (t < s) red[t] = fmaxf(red[t], red[t + s]); __syncthreads(); }
    float mx = red[0]; __syncthreads();
    float e = __expf(v - mx);
    red[t] = e; __syncthreads();
    for (int s = 128; s > 0; s >>= 1) { if (t < s) red[t] += red[t + s]; __syncthreads(); }
    float a = e / red[0];

    int c   = t >> 5, twc = t & 31;
    int sel = (twc >> 3) & 1;             // lane half
    int j   = (twc & 7) + 8 * (twc >> 4); // element
    long cb = (bl * 8 + c) * 32;
    attnA[(cb + (h + 16 * sel)) * 16 + j] = tobf(a);
    if (h == 0) {                         // zero the pad rows m = 8..15
        #pragma unroll
        for (int m2 = 8; m2 < 16; ++m2)
            attnA[(cb + (m2 + 16 * sel)) * 16 + j] = (bf16_t)0.0f;
    }
}

// ---------------------------------------------------------------------------
// Kernel 4: Vbar[b,l,h,:] = attn[b,l,h,:] @ V[b,l,:,:]   (streams V: 255 MB)
// 1 wave per (b,l, 64-wide d-tile); V staged through LDS as bf16 transpose
// so B-operand reads are contiguous ds_load_b128; 8 t-chunks x 4 WMMAs.
// ---------------------------------------------------------------------------
__global__ __launch_bounds__(32) void k_vbar(const float* __restrict__ V,
                                             const bf16_t* __restrict__ attnA,
                                             float* __restrict__ Vbar) {
    __shared__ bf16_t ldsB[64 * 32];           // [d_local][t_local] bf16
    int blk = blockIdx.x;
    long bl = blk / 18;
    int  dt = blk % 18;
    int  d0 = dt * 64;
    int lane = threadIdx.x;

    BF16x16 a[8];
    #pragma unroll
    for (int c = 0; c < 8; ++c) {
        const uint4* p = (const uint4*)(attnA + ((bl * 8 + c) * 32 + lane) * 16);
        a[c].q[0] = p[0]; a[c].q[1] = p[1];
    }
    v8f acc[4] = {};

    for (int c = 0; c < 8; ++c) {
        const float* vrow = V + (bl * T_ + c * 32 + lane) * D_ + d0;
        #pragma unroll
        for (int q = 0; q < 16; ++q) {
            float4 f = *(const float4*)(vrow + q * 4);
            ldsB[(q * 4 + 0) * 32 + lane] = tobf(f.x);
            ldsB[(q * 4 + 1) * 32 + lane] = tobf(f.y);
            ldsB[(q * 4 + 2) * 32 + lane] = tobf(f.z);
            ldsB[(q * 4 + 3) * 32 + lane] = tobf(f.w);
        }
        __syncthreads();
        int col = lane & 15, kr = (lane >> 4) * 8;
        #pragma unroll
        for (int n = 0; n < 4; ++n) {
            BF16x16 Bv;
            const uint4* bp = (const uint4*)&ldsB[(n * 16 + col) * 32 + kr];
            Bv.q[0] = bp[0]; Bv.q[1] = bp[2];
            acc[n] = __builtin_amdgcn_wmma_f32_16x16x32_bf16(false, a[c].v, false, Bv.v,
                                                             (short)0, acc[n], false, false);
        }
        __syncthreads();
    }
    int hi = lane >> 4, n16 = lane & 15;
    #pragma unroll
    for (int n = 0; n < 4; ++n)
        #pragma unroll
        for (int r = 0; r < 8; ++r) {
            int m = r + hi * 8;            // head row
            if (m < H_)
                Vbar[(bl * H_ + m) * D_ + d0 + n * 16 + n16] = acc[n][r];
        }
}

// ---------------------------------------------------------------------------
// Kernel 5a: pooled (head-dim-major flatten f = e*8 + h):
//   poolF[b,l, e*8+h] = Vbar[b,l,h,:] @ Wv[g][:, h*64+e] + bv[g, h*64+e]
// ---------------------------------------------------------------------------
__global__ __launch_bounds__(64) void k_pool(const float* __restrict__ Vbar,
                                             const float* __restrict__ Wv,
                                             const float* __restrict__ bv,
                                             float* __restrict__ poolF) {
    int blh = blockIdx.x;
    long bl = blh / H_;
    int  h  = blh % H_;
    int  e  = threadIdx.x;
    int  l  = (int)(bl % L_);
    int  g  = l / GS_;
    float acc = bv[g * DD_ + h * HD_ + e];
    const float* vb = Vbar + (bl * H_ + h) * D_;
    const float* w  = Wv + (long)g * D_ * DD_ + h * HD_ + e;
    for (int d = 0; d < D_; ++d) acc += vb[d] * w[(long)d * DD_];
    poolF[bl * DD_ + e * H_ + h] = acc;
}

// ---------------------------------------------------------------------------
// Kernel 5b: out = pooled @ Wo[g] + bo[g].  All 24 rows of a group staged in
// LDS (48 KB) so each Wo element is read from HBM exactly once.
// ---------------------------------------------------------------------------
__global__ __launch_bounds__(256) void k_out(const float* __restrict__ poolF,
                                             const float* __restrict__ Wo,
                                             const float* __restrict__ bo,
                                             float* __restrict__ out) {
    __shared__ float pl[24 * DD_];             // 48 KB
    int ot = blockIdx.x;                       // 0..7
    int g  = blockIdx.y;                       // 0..8
    int o  = ot * 256 + threadIdx.x;
    for (int i = threadIdx.x; i < 24 * DD_; i += 256) {
        int r = i >> 9, f = i & 511;
        int b = r / GS_, s = r % GS_;
        pl[i] = poolF[((long)(b * L_ + g * GS_ + s)) * DD_ + f];
    }
    __syncthreads();
    float acc[24];
    float bb = bo[g * OD_ + o];
    #pragma unroll
    for (int r = 0; r < 24; ++r) acc[r] = bb;
    const float* w = Wo + (long)g * DD_ * OD_ + o;
    for (int f = 0; f < DD_; ++f) {
        float wv = w[(long)f * OD_];
        #pragma unroll
        for (int r = 0; r < 24; ++r) acc[r] += pl[r * DD_ + f] * wv;
    }
    #pragma unroll
    for (int r = 0; r < 24; ++r) {
        int b = r / GS_, s = r % GS_;
        out[((long)(b * L_ + g * GS_ + s)) * OD_ + o] = acc[r];
    }
}

// ---------------------------------------------------------------------------
// Launch
// ---------------------------------------------------------------------------
extern "C" void kernel_launch(void* const* d_in, const int* in_sizes, int n_in,
                              void* d_out, int out_size, void* d_ws, size_t ws_size,
                              hipStream_t stream) {
    const float* K     = (const float*)d_in[0];
    const float* V     = (const float*)d_in[1];
    const float* query = (const float*)d_in[2];
    const float* Wq    = (const float*)d_in[3];
    const float* bq    = (const float*)d_in[4];
    const float* Wk    = (const float*)d_in[5];
    // bk (d_in[6]) intentionally unused: constant over t -> softmax-invariant.
    const float* Wv    = (const float*)d_in[7];
    const float* bv    = (const float*)d_in[8];
    const float* Wo    = (const float*)d_in[9];
    const float* bo    = (const float*)d_in[10];
    float* out = (float*)d_out;

    char* ws = (char*)d_ws;
    float*  Qp     = (float*)(ws);                       //    55,296 B
    bf16_t* qtB    = (bf16_t*)(ws + 55296);              //   995,328 B
    float*  scores = (float*)(ws + 1050624);             // 1,769,472 B
    bf16_t* attnA  = (bf16_t*)(ws + 2820096);            // 1,769,472 B
    float*  Vbar   = (float*)(ws + 4589568);             // 7,962,624 B
    float*  poolF  = (float*)(ws + 12552192);            //   442,368 B
    // total ws use: 12,994,560 B

    k_qproj  <<<54,               256, 0, stream>>>(query, Wq, bq, Qp);
    k_qtilde <<<1944,             256, 0, stream>>>(Qp, Wk, qtB);
    k_scores <<<B_ * L_ * 16,      32, 0, stream>>>(K, qtB, scores);
    k_softmax<<<B_ * L_ * H_,     256, 0, stream>>>(scores, attnA);
    k_vbar   <<<B_ * L_ * 18,      32, 0, stream>>>(V, attnA, Vbar);
    k_pool   <<<B_ * L_ * H_,      64, 0, stream>>>(Vbar, Wv, bv, poolF);
    k_out    <<<dim3(8, 9),       256, 0, stream>>>(poolF, Wo, bo, out);
}
